// LSTMRegress_82849919140528
// MI455X (gfx1250) — matrix operand order repaired and checked by reading, not compiled, not run on hardware
//
#include <hip/hip_runtime.h>
#include <hip/hip_bf16.h>

#define T_STEPS   1024
#define BATCH     128
#define EMBED     58
#define EMBED_PAD 64
#define HIDDEN    256
#define OUT_DIM   33
#define HSTRIDE   264   // padded LDS row stride (bf16 elems) to spread banks

typedef __attribute__((ext_vector_type(16))) __bf16 v16bf;
typedef __attribute__((ext_vector_type(8)))  float  v8f;

union Frag { v16bf v; uint4 q[2]; };

__device__ __forceinline__ v8f wmma_bf16(v16bf a, v16bf b, v8f c) {
  // D = A(16x32 bf16) x B(32x16 bf16) + C(16x16 f32)
  return __builtin_amdgcn_wmma_f32_16x16x32_bf16(false, a, false, b,
                                                 (short)0, c, false, false);
}

__device__ __forceinline__ float fast_sigmoid(float x) {
  return 1.0f / (1.0f + __expf(-x));
}
__device__ __forceinline__ float fast_tanh(float x) {
  float e = __expf(2.0f * x);
  return 1.0f - 2.0f / (e + 1.0f);
}

// ---------------------------------------------------------------------------
// Pack x (T,B,58) f32 -> (T,B,64) bf16, zero-padded K
// ---------------------------------------------------------------------------
__global__ void pack_x_kernel(const float* __restrict__ x, __bf16* __restrict__ dst) {
  long e = (long)blockIdx.x * blockDim.x + threadIdx.x;
  long n = (long)T_STEPS * BATCH * EMBED_PAD;
  if (e >= n) return;
  int k = (int)(e & (EMBED_PAD - 1));
  long tb = e >> 6;
  float v = (k < EMBED) ? x[tb * EMBED + k] : 0.0f;
  dst[e] = (__bf16)v;
}

// ---------------------------------------------------------------------------
// Pack combined [W_ih | W_hh] into WMMA B-fragment layout:
// dst[((nt*KT+kt)*32 + lane)*16 + v], lane<16 -> K_local 0..7,16..23, lane>=16 -> 8..15,24..31
// B element (k,n) = W[n,k]; rows = 1024 gate columns, NT = 64 tiles.
// ---------------------------------------------------------------------------
__global__ void pack_wcomb_kernel(const float* __restrict__ Wih,
                                  const float* __restrict__ Whh,
                                  __bf16* __restrict__ dst,
                                  int Kreal, int KinPad, int KT) {
  long e = (long)blockIdx.x * blockDim.x + threadIdx.x;
  long n = (long)64 * KT * 512;
  if (e >= n) return;
  int v    = (int)(e & 15);
  int lane = (int)((e >> 4) & 31);
  int ft   = (int)(e >> 9);            // nt*KT + kt
  int kt   = ft % KT;
  int nt   = ft / KT;
  int K = kt * 32 + ((lane >= 16) ? 8 : 0) + (v < 8 ? v : v + 8);
  int N = nt * 16 + (lane & 15);
  float val;
  if (K < KinPad) val = (K < Kreal) ? Wih[(long)N * Kreal + K] : 0.0f;
  else            val = Whh[(long)N * HIDDEN + (K - KinPad)];
  dst[e] = (__bf16)val;
}

// ---------------------------------------------------------------------------
// Pack W_out (33,256) into B fragments, N padded to 48 (NT=3, KT=8)
// ---------------------------------------------------------------------------
__global__ void pack_wout_kernel(const float* __restrict__ W, __bf16* __restrict__ dst) {
  long e = (long)blockIdx.x * blockDim.x + threadIdx.x;
  long n = (long)3 * 8 * 512;
  if (e >= n) return;
  int v    = (int)(e & 15);
  int lane = (int)((e >> 4) & 31);
  int ft   = (int)(e >> 9);
  int kt   = ft % 8;
  int nt   = ft / 8;
  int K = kt * 32 + ((lane >= 16) ? 8 : 0) + (v < 8 ? v : v + 8);
  int N = nt * 16 + (lane & 15);
  float val = (N < OUT_DIM) ? W[(long)N * HIDDEN + K] : 0.0f;
  dst[e] = (__bf16)val;
}

// ---------------------------------------------------------------------------
// Recurrent LSTM layer. One workgroup, 512 threads = 16 waves.
// Wave w owns hidden slice [w*16, w*16+16) and its 4 gate column-tiles.
// K tiles: KXT from x-source (global bf16, row stride KXT*32), 8 from h (LDS).
// h double-buffered in LDS -> one barrier per step. L1: fused out-projection.
// Weight fragments are re-streamed from L2 *per chunk*; the opaque offset is
// re-materialized inside the chunk loop so neither LICM (across t) nor CSE
// (across chunks) can widen fragment live ranges into spills.
// ---------------------------------------------------------------------------
template <int KXT, bool L1>
__global__ __launch_bounds__(512)
void lstm_layer_kernel(const __bf16* __restrict__ xsrc,
                       const __bf16* __restrict__ Wp,
                       const float*  __restrict__ bih,
                       const float*  __restrict__ bhh,
                       const float*  __restrict__ h0,
                       const float*  __restrict__ c0,
                       __bf16*       __restrict__ yout,   // layer0 only
                       const __bf16* __restrict__ wop,    // layer1 only
                       const float*  __restrict__ bout,   // layer1 only
                       float*        __restrict__ outp,   // layer1 only
                       float*        __restrict__ hnp,
                       float*        __restrict__ cnp) {
  constexpr int KT = KXT + HIDDEN / 32;   // total K tiles per step
  constexpr int XS = KXT * 32;            // x-source row stride (elements)
  __shared__ __bf16 hsh[2 * BATCH * HSTRIDE];   // 132 KB, double-buffered h

  const int tid    = threadIdx.x;
  const int lane   = tid & 31;
  const int wv     = tid >> 5;            // 0..15 == hidden slice
  const int lane15 = lane & 15;
  const int koff   = (lane & 16) ? 8 : 0; // A-frag K sub-offset per half-wave
  const int rbase  = (lane & 16) ? 8 : 0; // C-layout: lanes>=16 hold M=8..15
  const int colj   = wv * 16 + lane15;    // hidden column owned by this lane

  // --- init: h0 -> LDS buf0 (bf16), c0 -> registers, bias preload ---
  for (int i = tid; i < BATCH * HIDDEN; i += 512) {
    int b = i >> 8, j = i & (HIDDEN - 1);
    hsh[b * HSTRIDE + j] = (__bf16)h0[i];
  }
  float creg[8][8];
#pragma unroll
  for (int mt = 0; mt < 8; ++mt)
#pragma unroll
    for (int r = 0; r < 8; ++r)
      creg[mt][r] = c0[(mt * 16 + rbase + r) * HIDDEN + colj];
  float bi[4];
#pragma unroll
  for (int g = 0; g < 4; ++g)
    bi[g] = bih[g * HIDDEN + colj] + bhh[g * HIDDEN + colj];
  __syncthreads();

  int pb = 0;
  for (int t = 0; t < T_STEPS; ++t) {
    const __bf16* xrow = xsrc + (size_t)t * BATCH * XS;
    const __bf16* hcur = hsh + pb * (BATCH * HSTRIDE);
    __bf16*       hnew = hsh + (1 - pb) * (BATCH * HSTRIDE);

#pragma unroll
    for (int ch = 0; ch < 4; ++ch) {           // 2 M-tiles per chunk
      const int mt0 = ch * 2;

      // Opaque zero offset, fresh per chunk: B-fragment loads below can be
      // neither hoisted out of the time loop (LICM) nor merged across the
      // four chunks (CSE) -- either would pin 100s of VGPRs and spill.
      // Base pointer provenance is preserved -> still global_load_b128.
      size_t wo = 0;
      asm volatile("" : "+s"(wo));
      const __bf16* WpT = Wp + wo;

      v8f acc[2][4];
#pragma unroll
      for (int m = 0; m < 2; ++m)
#pragma unroll
        for (int g = 0; g < 4; ++g)
          acc[m][g] = v8f{0.f, 0.f, 0.f, 0.f, 0.f, 0.f, 0.f, 0.f};

#pragma unroll
      for (int kt = 0; kt < KT; ++kt) {
        Frag a[2];
#pragma unroll
        for (int m = 0; m < 2; ++m) {
          const int row = (mt0 + m) * 16 + lane15;
          if (kt < KXT) {
            const uint4* ap = (const uint4*)(xrow + (size_t)row * XS + kt * 32 + koff);
            a[m].q[0] = ap[0];      // K koff..koff+7
            a[m].q[1] = ap[2];      // K koff+16..koff+23
          } else {
            const uint4* ap = (const uint4*)(hcur + row * HSTRIDE + (kt - KXT) * 32 + koff);
            a[m].q[0] = ap[0];
            a[m].q[1] = ap[2];
          }
        }
        Frag bq[4];
#pragma unroll
        for (int g = 0; g < 4; ++g) {
          const uint4* bp =
              (const uint4*)(WpT + ((size_t)((g * 16 + wv) * KT + kt)) * 512 + lane * 16);
          bq[g].q[0] = bp[0];
          bq[g].q[1] = bp[1];
        }
#pragma unroll
        for (int m = 0; m < 2; ++m)
#pragma unroll
          for (int g = 0; g < 4; ++g)
            acc[m][g] = wmma_bf16(a[m].v, bq[g].v, acc[m][g]);
      }

      // --- elementwise LSTM cell update; all 4 gates are lane-local ---
#pragma unroll
      for (int m = 0; m < 2; ++m) {
        const int mt = mt0 + m;
#pragma unroll
        for (int r = 0; r < 8; ++r) {
          float iv = fast_sigmoid(acc[m][0][r] + bi[0]);
          float fv = fast_sigmoid(acc[m][1][r] + bi[1]);
          float gv = fast_tanh(acc[m][2][r] + bi[2]);
          float ov = fast_sigmoid(acc[m][3][r] + bi[3]);
          float cc = fv * creg[mt][r] + iv * gv;
          creg[mt][r] = cc;
          float hv = ov * fast_tanh(cc);
          const int brow = mt * 16 + rbase + r;
          __bf16 hb = (__bf16)hv;
          hnew[brow * HSTRIDE + colj] = hb;
          if (!L1) yout[((size_t)t * BATCH + brow) * HIDDEN + colj] = hb;
          if (t == T_STEPS - 1) {
            hnp[brow * HIDDEN + colj] = hv;
            cnp[brow * HIDDEN + colj] = cc;
          }
        }
      }
    }
    __syncthreads();   // all writes of h_t visible; all reads of h_{t-1} done

    if (L1) {
      // fused output projection + softplus on h_t (24 tiles over 16 waves)
      size_t wo2 = 0;
      asm volatile("" : "+s"(wo2));
      const __bf16* wopT = wop + wo2;
#pragma unroll
      for (int tb = 0; tb < 2; ++tb) {
        const int tile = wv + tb * 16;
        if (tile < 24) {
          const int mt = tile / 3, nt = tile % 3;
          v8f acc = v8f{0.f, 0.f, 0.f, 0.f, 0.f, 0.f, 0.f, 0.f};
#pragma unroll
          for (int kt = 0; kt < HIDDEN / 32; ++kt) {
            Frag a, b;
            const uint4* ap =
                (const uint4*)(hnew + (mt * 16 + lane15) * HSTRIDE + kt * 32 + koff);
            a.q[0] = ap[0];
            a.q[1] = ap[2];
            const uint4* bp = (const uint4*)(wopT + ((size_t)(nt * 8 + kt)) * 512 + lane * 16);
            b.q[0] = bp[0];
            b.q[1] = bp[1];
            acc = wmma_bf16(a.v, b.v, acc);
          }
          const int col = nt * 16 + lane15;
          if (col < OUT_DIM) {
            const float bb = bout[col];
#pragma unroll
            for (int r = 0; r < 8; ++r) {
              const int brow = mt * 16 + rbase + r;
              float v = acc[r] + bb;
              // stable softplus
              float sp = fmaxf(v, 0.0f) + __logf(1.0f + __expf(-fabsf(v)));
              outp[((size_t)t * BATCH + brow) * OUT_DIM + col] = sp;
            }
          }
        }
      }
    }
    pb ^= 1;
  }
}

// ---------------------------------------------------------------------------
extern "C" void kernel_launch(void* const* d_in, const int* in_sizes, int n_in,
                              void* d_out, int out_size, void* d_ws, size_t ws_size,
                              hipStream_t stream) {
  const float* x    = (const float*)d_in[0];
  const float* h0   = (const float*)d_in[1];
  const float* c0   = (const float*)d_in[2];
  const float* Wih0 = (const float*)d_in[3];
  const float* Whh0 = (const float*)d_in[4];
  const float* bih0 = (const float*)d_in[5];
  const float* bhh0 = (const float*)d_in[6];
  const float* Wih1 = (const float*)d_in[7];
  const float* Whh1 = (const float*)d_in[8];
  const float* bih1 = (const float*)d_in[9];
  const float* bhh1 = (const float*)d_in[10];
  const float* Wout = (const float*)d_in[11];
  const float* bout = (const float*)d_in[12];

  float* out = (float*)d_out;
  float* hn  = out + (size_t)T_STEPS * BATCH * OUT_DIM;
  float* cn  = hn + (size_t)2 * BATCH * HIDDEN;

  char* ws = (char*)d_ws;
  size_t off = 0;
  __bf16* xpk = (__bf16*)(ws + off); off += (size_t)T_STEPS * BATCH * EMBED_PAD * 2;
  __bf16* y0  = (__bf16*)(ws + off); off += (size_t)T_STEPS * BATCH * HIDDEN * 2;
  __bf16* w0p = (__bf16*)(ws + off); off += (size_t)64 * 10 * 512 * 2;
  __bf16* w1p = (__bf16*)(ws + off); off += (size_t)64 * 16 * 512 * 2;
  __bf16* wop = (__bf16*)(ws + off); off += (size_t)3 * 8 * 512 * 2;
  if (off > ws_size) return;

  {
    long n = (long)T_STEPS * BATCH * EMBED_PAD;
    pack_x_kernel<<<(n + 255) / 256, 256, 0, stream>>>(x, xpk);
  }
  {
    long n = (long)64 * 10 * 512;
    pack_wcomb_kernel<<<(n + 255) / 256, 256, 0, stream>>>(Wih0, Whh0, w0p,
                                                           EMBED, EMBED_PAD, 10);
  }
  {
    long n = (long)64 * 16 * 512;
    pack_wcomb_kernel<<<(n + 255) / 256, 256, 0, stream>>>(Wih1, Whh1, w1p,
                                                           HIDDEN, HIDDEN, 16);
  }
  {
    long n = (long)3 * 8 * 512;
    pack_wout_kernel<<<(n + 255) / 256, 256, 0, stream>>>(Wout, wop);
  }

  // layer 0: x(pad64) -> y0, writes hn[0], cn[0]
  lstm_layer_kernel<EMBED_PAD / 32, false><<<1, 512, 0, stream>>>(
      xpk, w0p, bih0, bhh0, h0, c0, y0,
      nullptr, nullptr, nullptr, hn, cn);

  // layer 1: y0 -> fused out-projection + softplus, writes hn[1], cn[1]
  lstm_layer_kernel<HIDDEN / 32, true><<<1, 512, 0, stream>>>(
      y0, w1p, bih1, bhh1, h0 + (size_t)BATCH * HIDDEN, c0 + (size_t)BATCH * HIDDEN,
      nullptr, wop, bout, out,
      hn + (size_t)BATCH * HIDDEN, cn + (size_t)BATCH * HIDDEN);
}
